// LegacyEPMoE_6365141532679
// MI455X (gfx1250) — matrix operand order repaired and checked by reading, not compiled, not run on hardware
//
#include <hip/hip_runtime.h>
#include <hip/hip_bf16.h>

// Problem constants (match reference: T,H,E,I,K = 1024,1024,8,2048,2)
#define T_TOK 1024
#define H_DIM 1024
#define E_NUM 8
#define I_DIM 2048

typedef __bf16 bf16_t;
typedef __attribute__((ext_vector_type(16))) __bf16 v16bf;
typedef __attribute__((ext_vector_type(8)))  __bf16 v8bf;
typedef __attribute__((ext_vector_type(8)))  float  v8f;
typedef __attribute__((ext_vector_type(4)))  float  v4f;

__device__ __forceinline__ int imin(int a, int b){ return a < b ? a : b; }

// Pack 16 f32 -> v16bf (order: a,b = low-K chunk pair, c,d = high-K chunk pair)
__device__ __forceinline__ v16bf pack_cvt(v4f a, v4f b, v4f c, v4f d){
  v16bf r;
#pragma unroll
  for (int i = 0; i < 4; ++i){
    r[i]      = (bf16_t)a[i];
    r[4 + i]  = (bf16_t)b[i];
    r[8 + i]  = (bf16_t)c[i];
    r[12 + i] = (bf16_t)d[i];
  }
  return r;
}

// ---------------------------------------------------------------------------
// 1) Router: softmax over 8 logits, top-2, normalized weights, scatter into
//    per-expert compact lists (counts/ids/wts live in workspace).
// ---------------------------------------------------------------------------
__global__ void moe_router(const float* __restrict__ logits,
                           int* __restrict__ counts,
                           int* __restrict__ ids,
                           float* __restrict__ wts){
  int t = blockIdx.x * blockDim.x + threadIdx.x;
  if (t >= T_TOK) return;
  float l[E_NUM];
  float mx = -3.4e38f;
#pragma unroll
  for (int e = 0; e < E_NUM; ++e){ l[e] = logits[t * E_NUM + e]; mx = fmaxf(mx, l[e]); }
#pragma unroll
  for (int e = 0; e < E_NUM; ++e){ l[e] = __expf(l[e] - mx); }
  // top-2 (lowest index wins ties, matching lax.top_k)
  int i1 = 0;
#pragma unroll
  for (int e = 1; e < E_NUM; ++e){ if (l[e] > l[i1]) i1 = e; }
  int i2 = (i1 == 0) ? 1 : 0;
#pragma unroll
  for (int e = 0; e < E_NUM; ++e){ if (e != i1 && l[e] > l[i2]) i2 = e; }
  float p1 = l[i1], p2 = l[i2];
  float inv = 1.0f / (p1 + p2);
  int s1 = atomicAdd(&counts[i1], 1);
  ids[i1 * T_TOK + s1] = t;  wts[i1 * T_TOK + s1] = p1 * inv;
  int s2 = atomicAdd(&counts[i2], 1);
  ids[i2 * T_TOK + s2] = t;  wts[i2 * T_TOK + s2] = p2 * inv;
}

// ---------------------------------------------------------------------------
// 2) Prefix sum of per-expert counts -> compact act-buffer offsets
// ---------------------------------------------------------------------------
__global__ void moe_prefix(const int* __restrict__ counts, int* __restrict__ offsets){
  if (threadIdx.x == 0 && blockIdx.x == 0){
    int a = 0;
#pragma unroll
    for (int e = 0; e < E_NUM; ++e){ offsets[e] = a; a += counts[e]; }
  }
}

// ---------------------------------------------------------------------------
// 3) Grouped GEMM1: act = silu(Xg * W13_gate^T) * (Xg * W13_up^T)  (bf16 out)
//    One wave -> 16 tokens x 64 cols; 4 gate + 4 up WMMA tiles per K-step.
//    grid = (2048/512, 64, 8), block = 256 (8 waves)
// ---------------------------------------------------------------------------
__global__ void __launch_bounds__(256)
moe_gemm1(const float* __restrict__ X, const float* __restrict__ W13,
          const int* __restrict__ counts, const int* __restrict__ offsets,
          const int* __restrict__ ids, bf16_t* __restrict__ act){
  const int e     = blockIdx.z;
  const int count = counts[e];
  const int mbase = blockIdx.y * 16;
  if (mbase >= count) return;                 // uniform across block: EXEC stays all-1
  const int lane  = threadIdx.x & 31;
  const int wave  = threadIdx.x >> 5;
  const int n0    = (blockIdx.x * 8 + wave) * 64;   // gate column base (0..2047)
  const int offs  = offsets[e];

  // A-fragment addressing (16-bit A 16x32, contiguous k order shared with B):
  // lane L: row m = L&15; 16 contiguous k at kb, kb = (L>>4)*16
  const int mrow  = lane & 15;
  const int tok   = ids[e * T_TOK + imin(mbase + mrow, count - 1)];
  const float* xrow = X + (size_t)tok * H_DIM;
  const int kb_a  = (lane >> 4) * 16;

  // B-fragment addressing: lane L: col n = L&15; 16 contiguous k at kb_b=(L>>4)*16
  const int nlane = lane & 15;
  const int kb_b  = (lane >> 4) * 16;
  const float* w13e = W13 + (size_t)e * (2 * I_DIM) * H_DIM;

  v8f accg[4] = {}, accu[4] = {};

  for (int k = 0; k < H_DIM; k += 32){
    const v4f* pa0 = (const v4f*)(xrow + k + kb_a);
    const v4f* pa1 = (const v4f*)(xrow + k + kb_a + 8);
    v16bf afrag = pack_cvt(pa0[0], pa0[1], pa1[0], pa1[1]);
#pragma unroll
    for (int s = 0; s < 4; ++s){
      const int og = n0 + s * 16 + nlane;                 // gate output col
      const float* bg = w13e + (size_t)og * H_DIM + k + kb_b;
      __builtin_prefetch(bg + 32, 0, 1);
      const v4f* qg = (const v4f*)bg;
      v16bf bfrag = pack_cvt(qg[0], qg[1], qg[2], qg[3]);
      accg[s] = __builtin_amdgcn_wmma_f32_16x16x32_bf16(
          false, afrag, false, bfrag, (short)0, accg[s], false, false);

      const float* bu = w13e + (size_t)(og + I_DIM) * H_DIM + k + kb_b;  // up col
      const v4f* qu = (const v4f*)bu;
      v16bf ufrag = pack_cvt(qu[0], qu[1], qu[2], qu[3]);
      accu[s] = __builtin_amdgcn_wmma_f32_16x16x32_bf16(
          false, afrag, false, ufrag, (short)0, accu[s], false, false);
    }
  }

  // Epilogue: silu(g)*u -> bf16 act (compact segment rows, mask padding rows)
#pragma unroll
  for (int s = 0; s < 4; ++s){
#pragma unroll
    for (int r = 0; r < 8; ++r){
      const int m  = r + 8 * (lane >> 4);   // C/D layout: lanes16-31 hold M=r+8
      const int sl = mbase + m;
      if (sl < count){
        float g = accg[s][r], u = accu[s][r];
        float a = (g / (1.0f + __expf(-g))) * u;
        act[(size_t)(offs + sl) * I_DIM + n0 + s * 16 + nlane] = (bf16_t)a;
      }
    }
  }
}

// ---------------------------------------------------------------------------
// 4) Grouped GEMM2: out[tok,:] += w * (act_e * W2[e]^T)
//    grid = (1024/512, 64, 8), block = 256 (8 waves)
// ---------------------------------------------------------------------------
__global__ void __launch_bounds__(256)
moe_gemm2(const bf16_t* __restrict__ act, const float* __restrict__ W2,
          const int* __restrict__ counts, const int* __restrict__ offsets,
          const int* __restrict__ ids, const float* __restrict__ wts,
          float* __restrict__ out){
  const int e     = blockIdx.z;
  const int count = counts[e];
  const int mbase = blockIdx.y * 16;
  if (mbase >= count) return;
  const int lane  = threadIdx.x & 31;
  const int wave  = threadIdx.x >> 5;
  const int n0    = (blockIdx.x * 8 + wave) * 64;   // hidden column base (0..1023)
  const int offs  = offsets[e];

  const int kb_a  = (lane >> 4) * 16;
  // act buffer is padded past 2048 rows, so straddling tiles read in-bounds garbage
  const bf16_t* arow = act + (size_t)(offs + mbase + (lane & 15)) * I_DIM;

  const int nlane = lane & 15;
  const int kb_b  = (lane >> 4) * 16;
  const float* w2e = W2 + (size_t)e * H_DIM * I_DIM;

  v8f acc[4] = {};

  for (int k = 0; k < I_DIM; k += 32){
    v8bf a0 = *(const v8bf*)(arow + k + kb_a);
    v8bf a1 = *(const v8bf*)(arow + k + kb_a + 8);
    v16bf afrag;
#pragma unroll
    for (int i = 0; i < 8; ++i){ afrag[i] = a0[i]; afrag[8 + i] = a1[i]; }
#pragma unroll
    for (int s = 0; s < 4; ++s){
      const int h = n0 + s * 16 + nlane;
      const float* bp = w2e + (size_t)h * I_DIM + k + kb_b;
      __builtin_prefetch(bp + 32, 0, 1);
      const v4f* q = (const v4f*)bp;
      v16bf bfrag = pack_cvt(q[0], q[1], q[2], q[3]);
      acc[s] = __builtin_amdgcn_wmma_f32_16x16x32_bf16(
          false, afrag, false, bfrag, (short)0, acc[s], false, false);
    }
  }

  // Weighted scatter-add into output
#pragma unroll
  for (int s = 0; s < 4; ++s){
#pragma unroll
    for (int r = 0; r < 8; ++r){
      const int m  = r + 8 * (lane >> 4);
      const int sl = mbase + m;
      if (sl < count){
        const int   tok = ids[e * T_TOK + sl];
        const float w   = wts[e * T_TOK + sl];
        atomicAdd(&out[(size_t)tok * H_DIM + n0 + s * 16 + nlane], w * acc[s][r]);
      }
    }
  }
}

// ---------------------------------------------------------------------------
// Host launcher
// ---------------------------------------------------------------------------
extern "C" void kernel_launch(void* const* d_in, const int* in_sizes, int n_in,
                              void* d_out, int out_size, void* d_ws, size_t ws_size,
                              hipStream_t stream){
  const float* hidden = (const float*)d_in[0];   // (T, H)
  const float* logits = (const float*)d_in[1];   // (T, E)
  const float* w13    = (const float*)d_in[2];   // (E, 2I, H)
  const float* w2     = (const float*)d_in[3];   // (E, H, I)
  float* out = (float*)d_out;                    // (T, H)

  // Workspace layout
  char* ws = (char*)d_ws;
  int*    counts  = (int*)ws;                    // 8 ints
  int*    offsets = (int*)(ws + 32);             // 8 ints
  int*    ids     = (int*)(ws + 64);             // 8*1024 ints  (32 KB)
  float*  wts     = (float*)(ws + 64 + 32768);   // 8*1024 f32   (32 KB)
  bf16_t* act     = (bf16_t*)(ws + 65792);       // (2048+16) x 2048 bf16 (~8.5 MB)

  hipMemsetAsync(d_ws, 0, 64, stream);                               // counts/offsets
  hipMemsetAsync(d_out, 0, (size_t)out_size * sizeof(float), stream); // accum target

  moe_router<<<(T_TOK + 255) / 256, 256, 0, stream>>>(logits, counts, ids, wts);
  moe_prefix<<<1, 32, 0, stream>>>(counts, offsets);

  // GEMM1: gate cols 2048 / (8 waves * 64) = 4 blocks in x; 64 m-tiles; 8 experts
  moe_gemm1<<<dim3(4, T_TOK / 16, E_NUM), 256, 0, stream>>>(
      hidden, w13, counts, offsets, ids, act);

  // GEMM2: hidden cols 1024 / 512 = 2 blocks in x
  moe_gemm2<<<dim3(2, T_TOK / 16, E_NUM), 256, 0, stream>>>(
      act, w2, counts, offsets, ids, wts, out);
}
